// AttentionBlock_35029753266518
// MI455X (gfx1250) — matrix operand order repaired and checked by reading, compiled, not run on hardware
//
#include <hip/hip_runtime.h>
#include <math.h>

// Problem constants (reference: B=2, L=2048, d=512, H=8, dh=64, dff=2048)
#define BATCH   2
#define L_SEQ   2048
#define D_MODEL 512
#define NHEAD   8
#define DH      64
#define DFF     2048
#define LMAX    39000

typedef __attribute__((ext_vector_type(16))) __bf16 v16bf;
typedef __attribute__((ext_vector_type(8)))  float  v8f;

union BfFrag {
    unsigned int   u[8];
    unsigned short h[16];
    v16bf          v;
};

// Hardware RNE conversion: float -> bf16 (single v_cvt op, pair may fuse to v_cvt_pk)
__device__ inline unsigned short f2bf(float f) {
    return __builtin_bit_cast(unsigned short, (__bf16)f);
}
__device__ inline unsigned int cvt2(float x, float y) {
    return (unsigned int)f2bf(x) | ((unsigned int)f2bf(y) << 16);
}

// DPP butterfly max over a 16-lane row (stays inside each wave32 half).
// dpp_ctrl must be an ICE at the builtin call -> template parameter.
template <int CTRL>
__device__ inline float fmax_dpp(float v) {
    int t = __builtin_amdgcn_update_dpp(0, __builtin_bit_cast(int, v), CTRL, 0xF, 0xF, true);
    return fmaxf(v, __builtin_bit_cast(float, t));
}
__device__ inline float rowmax16(float v) {
    v = fmax_dpp<0xB1>(v);    // quad_perm(1,0,3,2)
    v = fmax_dpp<0x4E>(v);    // quad_perm(2,3,0,1)
    v = fmax_dpp<0x141>(v);   // row_half_mirror
    v = fmax_dpp<0x140>(v);   // row_mirror
    return v;
}

// ---------------------------------------------------------------------------
// Tiled WMMA GEMM: C[M,N] = A[M,K] @ B[K,N] + bias  (+ReLU / +residual)
// block = 128 threads (4 waves), tile 64x64, K-step 32 (= WMMA K).
// Wave grid 2x2: each wave owns a 32x32 quadrant = 2x2 WMMA tiles.
// EPI: 0 = bias, 1 = bias+relu, 2 = bias+residual
// ---------------------------------------------------------------------------
template <int EPI>
__global__ __launch_bounds__(128)
void gemm_wmma(const float* __restrict__ A, const float* __restrict__ B,
               const float* __restrict__ bias, const float* __restrict__ res,
               float* __restrict__ C, int M, int N, int K)
{
    __shared__ unsigned int   As[64][16];   // bf16 pairs along K
    __shared__ unsigned short Bs[32][64];

    const int tid    = threadIdx.x;
    const int wave   = tid >> 5;
    const int lane   = tid & 31;
    const int lh     = lane & 15;
    const int hiHalf = (lane >= 16) ? 1 : 0;
    const int wm     = (wave & 1) * 32;     // wave row quadrant
    const int wn     = (wave >> 1) * 32;    // wave col quadrant

    const int m0 = blockIdx.y * 64;
    const int n0 = blockIdx.x * 64;

    v8f acc[2][2] = {{{}, {}}, {{}, {}}};

    for (int k0 = 0; k0 < K; k0 += 32) {
        if (k0 + 32 < K) {   // prefetch next K tile (global_prefetch_b8)
            __builtin_prefetch(A + (size_t)(m0 + (tid >> 3)) * K + k0 + 32 + (tid & 7) * 4, 0, 0);
            __builtin_prefetch(B + (size_t)(k0 + 32 + (tid >> 2)) * N + n0 + (tid & 3) * 16, 0, 0);
        }
        __syncthreads();
        // stage A tile 64x32 (fp32 -> bf16, packed b64 stores)
        {
            int row = tid >> 3;            // 0..15
            int cu  = (tid & 7) * 2;       // uint col 0..14
            for (int r = 0; r < 4; ++r) {
                const float4 f = *(const float4*)(A + (size_t)(m0 + row + r * 16) * K + k0 + cu * 2);
                uint2 p; p.x = cvt2(f.x, f.y); p.y = cvt2(f.z, f.w);
                *(uint2*)&As[row + r * 16][cu] = p;
            }
        }
        // stage B tile 32x64 (packed b64 stores)
        {
            int row = tid >> 2;            // 0..31
            int cb  = (tid & 3) * 16;
            for (int c = 0; c < 4; ++c) {
                const float4 f = *(const float4*)(B + (size_t)(k0 + row) * N + n0 + cb + c * 4);
                uint2 p; p.x = cvt2(f.x, f.y); p.y = cvt2(f.z, f.w);
                *(uint2*)&Bs[row][cb + c * 4] = p;
            }
        }
        __syncthreads();

        // A fragments (16x32 A-layout): row = lane&15, K split by lane half
        BfFrag af[2];
        for (int i = 0; i < 2; ++i) {
            const int arow = wm + i * 16 + lh;
            const int h4   = hiHalf ? 8 : 0;
            for (int p = 0; p < 8; ++p)
                af[i].u[p] = As[arow][h4 + p];
        }
        // B fragments (32x16 B-layout): col = lane&15, 16 consecutive K per half
        BfFrag bf[2];
        for (int j = 0; j < 2; ++j) {
            const int kbase = hiHalf ? 16 : 0;
            const int nn    = wn + j * 16 + lh;
            for (int p = 0; p < 8; ++p)
                bf[j].u[p] = (unsigned int)Bs[kbase + 2 * p][nn]
                           | ((unsigned int)Bs[kbase + 2 * p + 1][nn] << 16);
        }
        for (int i = 0; i < 2; ++i)
            for (int j = 0; j < 2; ++j)
                acc[i][j] = __builtin_amdgcn_wmma_f32_16x16x32_bf16(
                    false, af[i].v, false, bf[j].v, (short)0, acc[i][j], false, false);
    }

    // epilogue / writeback (C-layout: M = r + 8*half, N = lane&15)
    for (int i = 0; i < 2; ++i) {
        const int gmBase = m0 + wm + i * 16 + hiHalf * 8;
        for (int j = 0; j < 2; ++j) {
            const int gn = n0 + wn + j * 16 + lh;
            const float bv = bias[gn];
            for (int r = 0; r < 8; ++r) {
                const int gm = gmBase + r;
                float v = acc[i][j][r] + bv;
                if (EPI == 1) v = fmaxf(v, 0.0f);
                if (EPI == 2) v += res[(size_t)gm * N + gn];
                C[(size_t)gm * N + gn] = v;
            }
        }
    }
}

// ---------------------------------------------------------------------------
// Flash attention with relative position bias.
// block = 128 threads (4 waves); each block: one (b, h, 64-query group);
// each wave owns a 16-query tile. Streams 32-key blocks through LDS.
// Row max: DPP butterfly. Row sum: extra WMMA against an all-ones "V" tile
// (accumulator o[4] carries the online softmax denominator, rescaled with O).
// qkv layout per row (b*L + l): head h at cols [h*192, h*192+192): q|k|v.
// ---------------------------------------------------------------------------
__global__ __launch_bounds__(128)
void attn_kernel(const float* __restrict__ qkv, const float* __restrict__ pos_bias,
                 float* __restrict__ out)
{
    __shared__ float          biasw[2 * L_SEQ - 1];   // 4095 floats, window for head h
    __shared__ unsigned short Ks[32][DH];
    __shared__ unsigned short Vs[32][DH];
    __shared__ unsigned short Ps[4][16][32];          // per-wave P round-trip buffer

    const int tid    = threadIdx.x;
    const int wave   = tid >> 5;
    const int lane   = tid & 31;
    const int lh     = lane & 15;
    const int hiHalf = (lane >= 16) ? 1 : 0;

    const int bid = blockIdx.x;
    const int qg  = bid & 31;
    const int h   = (bid >> 5) & 7;
    const int b   = bid >> 8;

    // stage pos_bias window: biasw[t] = pos_bias[(t + LMAX - L_SEQ)][h]
    for (int t = tid; t < 2 * L_SEQ - 1; t += 128)
        biasw[t] = pos_bias[(size_t)(t + (LMAX - L_SEQ)) * NHEAD + h];

    // Q fragments: 16 rows, dh=64 -> two 16x32 A-fragments (bf16)
    const int    qrow0  = qg * 64 + wave * 16;
    const size_t qkvRow = (size_t)(b * L_SEQ + qrow0 + lh) * (3 * D_MODEL) + (size_t)h * (3 * DH);
    BfFrag qf[2];
    {
        const int koff = hiHalf ? 16 : 0;
        for (int ks = 0; ks < 2; ++ks)
            for (int p = 0; p < 8; ++p) {
                int c = ks * 32 + koff + 2 * p;
                float2 f = *(const float2*)(qkv + qkvRow + c);
                qf[ks].u[p] = cvt2(f.x, f.y);
            }
    }
    BfFrag ones;                       // constant all-ones B fragment (bf16 1.0)
    for (int p = 0; p < 8; ++p) ones.u[p] = 0x3F803F80u;

    float rmax[8];
    v8f   o[5] = {{}, {}, {}, {}, {}}; // o[0..3] = O tiles, o[4] = running row-sum tile
    for (int r = 0; r < 8; ++r) rmax[r] = -3.0e38f;
    const float scale = 0.125f;        // 1/sqrt(64)

    for (int kb = 0; kb < L_SEQ / 32; ++kb) {
        const int k0 = kb * 32;
        __syncthreads();
        // stage K,V block (32 keys x 64) as bf16, packed b64 stores
        {
            int row = tid >> 2;
            int cb  = (tid & 3) * 16;
            const float* kp = qkv + (size_t)(b * L_SEQ + k0 + row) * (3 * D_MODEL)
                                  + (size_t)h * (3 * DH) + DH;
            for (int c = 0; c < 4; ++c) {
                float4 f = *(const float4*)(kp + cb + c * 4);
                uint2 pk; pk.x = cvt2(f.x, f.y); pk.y = cvt2(f.z, f.w);
                *(uint2*)&Ks[row][cb + c * 4] = pk;
                float4 g = *(const float4*)(kp + DH + cb + c * 4);
                uint2 pv; pv.x = cvt2(g.x, g.y); pv.y = cvt2(g.z, g.w);
                *(uint2*)&Vs[row][cb + c * 4] = pv;
            }
        }
        __syncthreads();

        // S = Q @ K^T : two 16x16 tiles (keys [k0,k0+16) and [k0+16,k0+32))
        v8f s[2] = {{}, {}};
        for (int nt = 0; nt < 2; ++nt) {
            const int n = nt * 16 + lh;
            for (int ks = 0; ks < 2; ++ks) {
                BfFrag kf;
                const int kk0 = ks * 32 + (hiHalf ? 16 : 0);
                for (int p = 0; p < 8; ++p)
                    kf.u[p] = *(const unsigned int*)&Ks[n][kk0 + 2 * p];
                s[nt] = __builtin_amdgcn_wmma_f32_16x16x32_bf16(
                    false, qf[ks].v, false, kf.v, (short)0, s[nt], false, false);
            }
        }

        // online softmax on (s + bias) * scale ; max via DPP, sum via o[4] WMMA
        const int qiBase = qrow0 + hiHalf * 8;
        float p0a[8], p1a[8];
        for (int r = 0; r < 8; ++r) {
            const int d0 = (qiBase + r) - (k0 + lh)      + (L_SEQ - 1);
            const int d1 = (qiBase + r) - (k0 + 16 + lh) + (L_SEQ - 1);
            float sv0 = (s[0][r] + biasw[d0]) * scale;
            float sv1 = (s[1][r] + biasw[d1]) * scale;
            float m = rowmax16(fmaxf(sv0, sv1));          // max over 16-lane half
            const float mnew = fmaxf(rmax[r], m);
            const float corr = __expf(rmax[r] - mnew);
            rmax[r] = mnew;
            p0a[r] = __expf(sv0 - mnew);
            p1a[r] = __expf(sv1 - mnew);
            for (int ot = 0; ot < 5; ++ot) o[ot][r] *= corr;
        }

        // P (C-layout) -> LDS -> A-layout fragment
        for (int r = 0; r < 8; ++r) {
            Ps[wave][hiHalf * 8 + r][lh]      = f2bf(p0a[r]);
            Ps[wave][hiHalf * 8 + r][16 + lh] = f2bf(p1a[r]);
        }
        asm volatile("s_wait_dscnt 0" ::: "memory");   // same-wave LDS in-order: stores visible
        BfFrag pf;
        {
            const int koff = hiHalf ? 16 : 0;
            for (int p = 0; p < 8; ++p) {
                int c = koff + 2 * p;
                pf.u[p] = *(const unsigned int*)&Ps[wave][lh][c];
            }
        }
        // O += P @ V (four dh tiles) ; o[4] += P @ ones (row sums)
        const int kkb = hiHalf ? 16 : 0;
        for (int ot = 0; ot < 4; ++ot) {
            BfFrag vf;
            const int n = ot * 16 + lh;
            for (int p = 0; p < 8; ++p)
                vf.u[p] = (unsigned int)Vs[kkb + 2 * p][n]
                        | ((unsigned int)Vs[kkb + 2 * p + 1][n] << 16);
            o[ot] = __builtin_amdgcn_wmma_f32_16x16x32_bf16(
                false, pf.v, false, vf.v, (short)0, o[ot], false, false);
        }
        o[4] = __builtin_amdgcn_wmma_f32_16x16x32_bf16(
            false, pf.v, false, ones.v, (short)0, o[4], false, false);
    }

    // write O / l  ->  attn_out[(b*L + q), h*64 + col]
    for (int ot = 0; ot < 4; ++ot)
        for (int r = 0; r < 8; ++r) {
            const int gm = qrow0 + hiHalf * 8 + r;
            const int gn = h * DH + ot * 16 + lh;
            out[(size_t)(b * L_SEQ + gm) * D_MODEL + gn] = o[ot][r] / o[4][r];
        }
}

// ---------------------------------------------------------------------------
// LayerNorm over last dim (d=512). One block per row, 256 threads.
// ---------------------------------------------------------------------------
__global__ __launch_bounds__(256)
void ln_kernel(const float* __restrict__ in, const float* __restrict__ gamma,
               const float* __restrict__ beta, float* __restrict__ out, int d)
{
    __shared__ float redS[8], redS2[8];
    __shared__ float muS, rsigS;
    const int row = blockIdx.x;
    const float* p = in + (size_t)row * d;

    float s = 0.0f, s2 = 0.0f;
    for (int i = threadIdx.x; i < d; i += 256) {
        float v = p[i]; s += v; s2 += v * v;
    }
    for (int m = 16; m >= 1; m >>= 1) {
        s  += __shfl_xor(s,  m, 32);
        s2 += __shfl_xor(s2, m, 32);
    }
    if ((threadIdx.x & 31) == 0) {
        redS[threadIdx.x >> 5] = s;
        redS2[threadIdx.x >> 5] = s2;
    }
    __syncthreads();
    if (threadIdx.x == 0) {
        float ts = 0.0f, ts2 = 0.0f;
        for (int i = 0; i < 8; ++i) { ts += redS[i]; ts2 += redS2[i]; }
        float mu  = ts / (float)d;
        float var = ts2 / (float)d - mu * mu;
        muS   = mu;
        rsigS = rsqrtf(var + 1e-5f);
    }
    __syncthreads();
    const float mu = muS, rs = rsigS;
    for (int i = threadIdx.x; i < d; i += 256)
        out[(size_t)row * d + i] = (p[i] - mu) * rs * gamma[i] + beta[i];
}

// ---------------------------------------------------------------------------
extern "C" void kernel_launch(void* const* d_in, const int* in_sizes, int n_in,
                              void* d_out, int out_size, void* d_ws, size_t ws_size,
                              hipStream_t stream)
{
    (void)in_sizes; (void)n_in; (void)out_size; (void)ws_size;
    const float* x        = (const float*)d_in[0];
    const float* Wqkv     = (const float*)d_in[1];
    const float* bqkv     = (const float*)d_in[2];
    const float* Wo       = (const float*)d_in[3];
    const float* bo       = (const float*)d_in[4];
    const float* pos_bias = (const float*)d_in[5];
    const float* W1       = (const float*)d_in[6];
    const float* b1       = (const float*)d_in[7];
    const float* W2       = (const float*)d_in[8];
    const float* b2       = (const float*)d_in[9];
    const float* gamma1   = (const float*)d_in[10];
    const float* beta1    = (const float*)d_in[11];
    const float* gamma2   = (const float*)d_in[12];
    const float* beta2    = (const float*)d_in[13];
    float* out = (float*)d_out;

    const int M = BATCH * L_SEQ;   // 4096
    float* ws   = (float*)d_ws;
    float* qkv  = ws; ws += (size_t)M * 3 * D_MODEL;   // 4096 x 1536
    float* attn = ws; ws += (size_t)M * D_MODEL;       // 4096 x 512
    float* y1   = ws; ws += (size_t)M * D_MODEL;
    float* hbuf = ws; ws += (size_t)M * D_MODEL;
    float* f1   = ws; ws += (size_t)M * DFF;           // 4096 x 2048
    float* y2   = ws;                                   // 4096 x 512

    // 1) qkv = x @ Wqkv + bqkv
    gemm_wmma<0><<<dim3((3 * D_MODEL) / 64, M / 64), 128, 0, stream>>>(
        x, Wqkv, bqkv, nullptr, qkv, M, 3 * D_MODEL, D_MODEL);
    // 2) flash attention with relative position bias
    attn_kernel<<<dim3(BATCH * NHEAD * (L_SEQ / 64)), 128, 0, stream>>>(qkv, pos_bias, attn);
    // 3) y1 = attn @ Wo + bo + x
    gemm_wmma<2><<<dim3(D_MODEL / 64, M / 64), 128, 0, stream>>>(
        attn, Wo, bo, x, y1, M, D_MODEL, D_MODEL);
    // 4) h = LN(y1)
    ln_kernel<<<dim3(M), 256, 0, stream>>>(y1, gamma1, beta1, hbuf, D_MODEL);
    // 5) f1 = relu(h @ W1 + b1)
    gemm_wmma<1><<<dim3(DFF / 64, M / 64), 128, 0, stream>>>(
        hbuf, W1, b1, nullptr, f1, M, DFF, D_MODEL);
    // 6) y2 = f1 @ W2 + b2 + h
    gemm_wmma<2><<<dim3(D_MODEL / 64, M / 64), 128, 0, stream>>>(
        f1, W2, b2, hbuf, y2, M, D_MODEL, DFF);
    // 7) out = LN(y2)
    ln_kernel<<<dim3(M), 256, 0, stream>>>(y2, gamma2, beta2, out, D_MODEL);
}
